// SimpleGNNBlock_75496935129487
// MI455X (gfx1250) — compile-verified, matrix operand order checked by reading, not run on hardware
//
#include <hip/hip_runtime.h>
#include <float.h>

// ---------------------------------------------------------------------------
// Fused GNN block for MI455X (gfx1250, wave32).
//   out[b,:] = sum_p relu( relu(feats(b,p) @ W1 + b1) @ W2 + b2 )
// Transposed formulation: per wave, N = 16 batch columns (16 b's at same p).
//   layer1: hT = W1^T(64x4) @ featsT(4x16)    -> 4x v_wmma_f32_16x16x4_f32
//   layer2: msgT = W2^T(64x64) @ hT(64x16)    -> 8x v_wmma_f32_16x16x32_f16
// D1 (C-layout: lane=batch col, vgpr=feature) feeds layer2's B operand
// directly after packed f32->f16 conversion — no cross-lane permutes.
// Round-3 tuning: v_med3_f32(x,0,FLT_MAX) relu in the accumulate path (the
// FLT_MAX clamp blocks the maxnum re-fold), WGP-scope prefetch via inline asm.
// ---------------------------------------------------------------------------

typedef float    v2f  __attribute__((ext_vector_type(2)));
typedef float    v8f  __attribute__((ext_vector_type(8)));
typedef _Float16 v16h __attribute__((ext_vector_type(16)));

#define B_TOT 32768
#define P_CNT 128
#define H_DIM 64

__global__ __launch_bounds__(256) void gnn_fused_wmma(
    const float* __restrict__ planet_xy,   // (B, P, 2)
    const float* __restrict__ planet_m,    // (P,)
    const float* __restrict__ ast_xy,      // (B, 2)
    const float* __restrict__ W1,          // (4, 64) row-major
    const float* __restrict__ b1,          // (64,)
    const float* __restrict__ W2,          // (64, 64) row-major
    const float* __restrict__ b2,          // (64,)
    float* __restrict__ out)               // (B, 64)
{
    const int lane = threadIdx.x & 31;
    const int wid  = threadIdx.x >> 5;
    const int col  = lane & 15;            // batch column within the 16-wide tile
    const int hi   = lane >> 4;            // 0: lanes 0-15, 1: lanes 16-31
    const int b    = ((blockIdx.x << 3) + wid) * 16 + col;

    // ---- A operand for layer1: W1^T tiles (16 features x K=4, f32) --------
    // A layout (32-bit 16x4): lanes 0-15 hold K=0 (v0), K=1 (v1);
    //                         lanes 16-31 hold K=2 (v0), K=3 (v1).
    v2f a1[4];
    {
        const int k0 = hi ? 2 : 0;
        #pragma unroll
        for (int t = 0; t < 4; ++t) {
            const int j = 16 * t + col;            // output feature (M row)
            a1[t].x = W1[(k0 + 0) * H_DIM + j];    // W1^T[j][k0]   = W1[k0][j]
            a1[t].y = W1[(k0 + 1) * H_DIM + j];
        }
    }

    // ---- biases in C layout (lane half selects M rows r+8) ----------------
    v8f vb1[4], vb2[4];
    #pragma unroll
    for (int t = 0; t < 4; ++t)
        #pragma unroll
        for (int r = 0; r < 8; ++r) {
            vb1[t][r] = b1[16 * t + r + 8 * hi];
            vb2[t][r] = b2[16 * t + r + 8 * hi];
        }

    // ---- A operand for layer2: W2^T tiles (16 out-feat x K=32, f16) -------
    // A layout (16-bit 16x32): element e of v16h holds
    //   K = (e&8)*2 + (hi?8:0) + (e&7)   within the 32-wide K chunk.
    v16h a2[4][2];
    #pragma unroll
    for (int to = 0; to < 4; ++to) {
        const int mo = 16 * to + col;              // output feature (M row)
        #pragma unroll
        for (int kc = 0; kc < 2; ++kc)
            #pragma unroll
            for (int e = 0; e < 16; ++e) {
                const int K = ((e & 8) << 1) + (hi ? 8 : 0) + (e & 7);
                const int k = 32 * kc + K;         // input feature index
                a2[to][kc][e] = (_Float16)W2[k * H_DIM + mo]; // W2^T[mo][k]
            }
    }

    // ---- per-lane asteroid position (fixed over the whole p loop) ---------
    const float ax = ast_xy[2 * b + 0];
    const float ay = ast_xy[2 * b + 1];

    // ---- accumulators: msgT summed over p, C layout -----------------------
    v8f acc[4];
    #pragma unroll
    for (int t = 0; t < 4; ++t)
        #pragma unroll
        for (int r = 0; r < 8; ++r) acc[t][r] = 0.0f;

    const float2* __restrict__ pxy = (const float2*)planet_xy;
    const long rowbase = (long)b * P_CNT;
    const v16h hzero = (v16h)(_Float16)0.0f;

    // Software pipeline: planet position + mass loaded one iteration ahead.
    float2 pl  = pxy[rowbase];
    float  mpv = planet_m[0];

    for (int p = 0; p < P_CNT; ++p) {
        const float2 cur  = pl;
        const float  mcur = mpv;
        const int    pn   = (p + 1 < P_CNT) ? (p + 1) : p;
        pl  = pxy[rowbase + pn];                   // next iteration's data
        mpv = planet_m[pn];                        // wave-uniform -> s_load

        // Prefetch the planet stream ~16 iterations ahead at WGP scope
        // (scope 0 pulls into all cache levels; the builtin can only reach
        // SE/DEV scope, so use inline asm with the default scope).
        {
            const int pp = (p + 16 < P_CNT) ? (p + 16) : (P_CNT - 1);
            const float2* pf = &pxy[rowbase + pp];
            asm volatile("global_prefetch_b8 %0, off" :: "v"(pf));
        }

        // ---- features for this (b, p) ------------------------------------
        const float dx  = cur.x - ax;
        const float dy  = cur.y - ay;
        const float inv = rsqrtf(dx * dx + dy * dy + 1e-6f);

        // B operand of layer1 (f32 4x16): lanes 0-15 carry (dx,dy) = K0,K1;
        // lanes 16-31 carry (invdist,m) = K2,K3.
        v2f fB;
        fB.x = hi ? inv  : dx;
        fB.y = hi ? mcur : dy;

        // ---- layer1: 4x v_wmma_f32_16x16x4_f32, C = bias1 ----------------
        v8f d1[4];
        #pragma unroll
        for (int t = 0; t < 4; ++t)
            d1[t] = __builtin_amdgcn_wmma_f32_16x16x4_f32(
                false, a1[t], false, fB, (short)0, vb1[t], false, false);

        // ---- f32->f16 (v_cvt_pk_f16_f32), then relu as v_pk_max_num_f16 --
        // relu commutes with the monotonic sign-preserving f16 conversion.
        // hlo covers K=0..31 (tiles 0,1), hhi covers K=32..63 (tiles 2,3).
        v16h hlo, hhi;
        #pragma unroll
        for (int j = 0; j < 8; ++j) {
            hlo[j]     = (_Float16)d1[0][j];
            hlo[j + 8] = (_Float16)d1[1][j];
            hhi[j]     = (_Float16)d1[2][j];
            hhi[j + 8] = (_Float16)d1[3][j];
        }
        hlo = __builtin_elementwise_max(hlo, hzero);
        hhi = __builtin_elementwise_max(hhi, hzero);

        // ---- layer2: 8x v_wmma_f32_16x16x32_f16, C = bias2 then chain ----
        #pragma unroll
        for (int to = 0; to < 4; ++to) {
            v8f m2 = __builtin_amdgcn_wmma_f32_16x16x32_f16(
                false, a2[to][0], false, hlo, (short)0, vb2[to], false, false);
            m2 = __builtin_amdgcn_wmma_f32_16x16x32_f16(
                false, a2[to][1], false, hhi, (short)0, m2, false, false);
            // relu via a single v_med3_f32(x, 0, FLT_MAX): the FLT_MAX upper
            // clamp keeps InstCombine from re-folding this into 2-op maxnum.
            #pragma unroll
            for (int j = 0; j < 8; ++j)
                acc[to][j] += __builtin_amdgcn_fmed3f(m2[j], 0.0f, FLT_MAX);
        }
    }

    // ---- store: lane's 8 floats are 8 contiguous output features ---------
    float* op = out + (long)b * H_DIM + 8 * hi;
    #pragma unroll
    for (int to = 0; to < 4; ++to)
        #pragma unroll
        for (int j = 0; j < 8; ++j)
            op[16 * to + j] = acc[to][j];          // merges into b128 stores
}

extern "C" void kernel_launch(void* const* d_in, const int* in_sizes, int n_in,
                              void* d_out, int out_size, void* d_ws, size_t ws_size,
                              hipStream_t stream) {
    (void)in_sizes; (void)n_in; (void)d_ws; (void)ws_size; (void)out_size;
    const float* planet_xy = (const float*)d_in[0];
    const float* planet_m  = (const float*)d_in[1];
    const float* ast_xy    = (const float*)d_in[2];
    const float* W1        = (const float*)d_in[3];
    const float* b1        = (const float*)d_in[4];
    const float* W2        = (const float*)d_in[5];
    const float* b2        = (const float*)d_in[6];
    float* out             = (float*)d_out;

    // 256 threads = 8 waves/block, 16 b per wave -> 128 b per block.
    dim3 grid(B_TOT / 128), block(256);
    gnn_fused_wmma<<<grid, block, 0, stream>>>(planet_xy, planet_m, ast_xy,
                                               W1, b1, W2, b2, out);
}
